// Network_4664334483617
// MI455X (gfx1250) — matrix operand order repaired and checked by reading, not compiled, not run on hardware
//
#include <hip/hip_runtime.h>
#include <hip/hip_bf16.h>

typedef __attribute__((ext_vector_type(16))) _Float16     v16h;
typedef __attribute__((ext_vector_type(8)))  float        v8f;
typedef __attribute__((ext_vector_type(4)))  unsigned int u4;

#define NR  512     // rays
#define NSC 64      // coarse samples
#define NSF 128     // fine samples (64 + 64 importance)
#define LD  320     // LDS activation row stride (halves)

// Canonical layer order: 0..7 hidden, 8 alpha, 9 feat, 10 view, 11 rgb
constexpr int KPAD_CE[12] = {64,256,256,256,256,320,256,256,256,256,288,128};
constexpr int NPAD_CE[12] = {256,256,256,256,256,256,256,256, 16,256,128, 16};
constexpr int KSRC_CE[12] = {63,256,256,256,256,319,256,256,256,256,283,128};
constexpr int NSRC_CE[12] = {256,256,256,256,256,256,256,256,  1,256,128,  3};
constexpr int WOFF_CE[12] = {0,16384,81920,147456,212992,278528,360448,425984,
                             491520,495616,561152,598016};
constexpr int BOFF_CE[12] = {600064,600320,600576,600832,601088,601344,601600,601856,
                             602112,602128,602384,602512};
constexpr int PARAM_HALVES = 602560;   // per-net fp16 param block (padded)

// ---------------------------------------------------------------------------
// Fragment loaders.
// A (16x32 f16, contiguous): lane<16 holds K 0..15, lane>=16 holds K 16..31
//   -> halves 0..15 = p[0..15] contiguous   (p at +0 / +16 per half)
// B (32x16 f16, contiguous, same k order as A): lane<16 holds K 0..15, lane>=16 holds K 16..31
//   -> halves 0..15 = p[0..15] contiguous
// ---------------------------------------------------------------------------
union H16 { v16h v; u4 q[2]; };

__device__ __forceinline__ v16h ld_fragA(const _Float16* p) {
  H16 h;
  h.q[0] = *(const u4*)(p);
  h.q[1] = *(const u4*)(p + 8);
  return h.v;
}
__device__ __forceinline__ v16h ld_fragB(const _Float16* p) {
  H16 h;
  h.q[0] = *(const u4*)(p);
  h.q[1] = *(const u4*)(p + 8);
  return h.v;
}

// ---------------------------------------------------------------------------
// One MLP layer: out[pts 16][N NT*16] = act( in[16][K] @ Wt^T + b )
// Executed per-wave; in/out are LDS activation buffers (row stride LD halves).
// Per K-step: batch all NT B-fragment loads (one clause, one wait), then run
// NT WMMAs back-to-back (no D->A/B hazards, no per-tile round trips).
// ---------------------------------------------------------------------------
template<int LI, int NT, bool RELU>
__device__ __forceinline__ void mlp_layer(const _Float16* __restrict__ in,  int inOff,
                                          _Float16* __restrict__ out,       int outOff,
                                          const _Float16* __restrict__ prm,
                                          int nbase_t, int lane, int pt0)
{
  constexpr int K  = KPAD_CE[LI];
  constexpr int NK = K / 32;
  const _Float16* bs = prm + BOFF_CE[LI];

  const int m   = lane & 15;
  const int koA = (lane & 16) ? 16 : 0;
  const int koB = (lane & 16) ? 16 : 0;
  const _Float16* arow  = in + (pt0 + m) * LD + inOff + koA;
  const _Float16* wbase = prm + WOFF_CE[LI] + (nbase_t * 16 + m) * K + koB;

  v8f acc[NT];
#pragma unroll
  for (int t = 0; t < NT; ++t)
#pragma unroll
    for (int r = 0; r < 8; ++r) acc[t][r] = 0.0f;

#pragma unroll
  for (int kk = 0; kk < NK; ++kk) {
    v16h a = ld_fragA(arow + kk * 32);
    v16h bf[NT];
#pragma unroll
    for (int t = 0; t < NT; ++t)
      bf[t] = ld_fragB(wbase + t * (16 * K) + kk * 32);   // const offsets -> imm
#pragma unroll
    for (int t = 0; t < NT; ++t)
      acc[t] = __builtin_amdgcn_wmma_f32_16x16x32_f16(
          false, a, false, bf[t], (short)0, acc[t], false, false);
  }

  const int mb = (lane & 16) ? 8 : 0;
#pragma unroll
  for (int t = 0; t < NT; ++t) {
    int n = (nbase_t + t) * 16 + m;
    float bb = (float)bs[n];
#pragma unroll
    for (int r = 0; r < 8; ++r) {
      float v = acc[t][r] + bb;
      if (RELU) v = fmaxf(v, 0.0f);
      out[(pt0 + mb + r) * LD + outOff + n] = (_Float16)v;
    }
  }
}

// Head: single padded WMMA tile, write VN valid columns to raw[P][4] at COL.
template<int LI, int VN, int COL>
__device__ __forceinline__ void mlp_head(const _Float16* __restrict__ in, int inOff,
                                         const _Float16* __restrict__ prm,
                                         float* __restrict__ raw, int Pbase,
                                         int lane, int pt0)
{
  constexpr int K  = KPAD_CE[LI];
  constexpr int NK = K / 32;
  const _Float16* bs = prm + BOFF_CE[LI];

  const int m   = lane & 15;
  const int koA = (lane & 16) ? 16 : 0;
  const int koB = (lane & 16) ? 16 : 0;
  const _Float16* arow  = in + (pt0 + m) * LD + inOff + koA;
  const _Float16* wbase = prm + WOFF_CE[LI] + m * K + koB;

  v8f acc;
#pragma unroll
  for (int r = 0; r < 8; ++r) acc[r] = 0.0f;

#pragma unroll
  for (int kk = 0; kk < NK; ++kk) {
    v16h a = ld_fragA(arow + kk * 32);
    v16h b = ld_fragB(wbase + kk * 32);
    acc = __builtin_amdgcn_wmma_f32_16x16x32_f16(
        false, a, false, b, (short)0, acc, false, false);
  }

  const int mb = (lane & 16) ? 8 : 0;
  if (m < VN) {
    float bb = (float)bs[m];
#pragma unroll
    for (int r = 0; r < 8; ++r)
      raw[(Pbase + pt0 + mb + r) * 4 + COL + m] = acc[r] + bb;
  }
}

// ---------------------------------------------------------------------------
// Fused NeRF MLP: 32 points per workgroup, 4 waves. Produces raw[P][4]=rgb,sigma
// ---------------------------------------------------------------------------
__global__ __launch_bounds__(128)
void nerf_mlp_kernel(const float* __restrict__ rays, const _Float16* __restrict__ prm,
                     const float* __restrict__ zsrc, int nsamp, float* __restrict__ raw)
{
  __shared__ _Float16 A[32 * LD];
  __shared__ _Float16 B[32 * LD];
  __shared__ _Float16 DE[32 * 32];

  const int tid  = threadIdx.x;
  const int lane = tid & 31;
  const int wave = tid >> 5;
  const int half = wave & 1;
  const int pt0  = (wave >> 1) * 16;
  const int Pbase = blockIdx.x * 32;

  // ---- build positional (pe, 63->64) and directional (de, 27->32) encodings
  {
    const int pt = tid >> 2, q = tid & 3;
    const int P = Pbase + pt;
    const int r = P / nsamp, s = P - r * nsamp;
    const float* rp = rays + r * 6;
    float d0 = rp[3], d1 = rp[4], d2 = rp[5];
    float scale = sqrtf(d0*d0 + d1*d1 + d2*d2);
    float z = zsrc ? zsrc[r * nsamp + s] : (100.0f * (float)s / 63.0f);
    float t = z / scale * 0.5f;               // xyz = (o + d*z/scale) / DIST
    float xyz[3] = { rp[0]*0.5f + d0*t, rp[1]*0.5f + d1*t, rp[2]*0.5f + d2*t };
    float dir[3] = { d0, d1, d2 };
#pragma unroll
    for (int ii = 0; ii < 16; ++ii) {
      int f = q * 16 + ii;
      float v = 0.0f;
      if (f < 3) v = xyz[f];
      else if (f < 63) {
        int e = f - 3, fr = e / 6, rm = e % 6;
        float ang = xyz[rm % 3] * (float)(1 << fr);
        v = (rm < 3) ? __sinf(ang) : __cosf(ang);
      }
      A[pt * LD + f] = (_Float16)v;
      B[pt * LD + f] = (_Float16)v;
    }
    if (q < 2) {
#pragma unroll
      for (int ii = 0; ii < 16; ++ii) {
        int f = q * 16 + ii;
        float v = 0.0f;
        if (f < 3) v = dir[f];
        else if (f < 27) {
          int e = f - 3, fr = e / 6, rm = e % 6;
          float ang = dir[rm % 3] * (float)(1 << fr);
          v = (rm < 3) ? __sinf(ang) : __cosf(ang);
        }
        DE[pt * 32 + f] = (_Float16)v;
      }
    }
  }
  __syncthreads();

  // Hidden trunk (pe stays at cols 0..63 of both buffers -> skip concat is free;
  // layer-5 weights were remapped for the zero pad at col 63).
  mlp_layer<0, 8, true >(A, 0,  B, 64, prm, half * 8, lane, pt0); __syncthreads();
  mlp_layer<1, 8, true >(B, 64, A, 64, prm, half * 8, lane, pt0); __syncthreads();
  mlp_layer<2, 8, true >(A, 64, B, 64, prm, half * 8, lane, pt0); __syncthreads();
  mlp_layer<3, 8, true >(B, 64, A, 64, prm, half * 8, lane, pt0); __syncthreads();
  mlp_layer<4, 8, true >(A, 64, B, 64, prm, half * 8, lane, pt0); __syncthreads();
  mlp_layer<5, 8, true >(B, 0,  A, 64, prm, half * 8, lane, pt0); __syncthreads(); // K=320 (skip)
  mlp_layer<6, 8, true >(A, 64, B, 64, prm, half * 8, lane, pt0); __syncthreads();
  mlp_layer<7, 8, true >(B, 64, A, 64, prm, half * 8, lane, pt0); __syncthreads();

  if (half == 0) mlp_head<8, 1, 3>(A, 64, prm, raw, Pbase, lane, pt0);   // sigma
  mlp_layer<9, 8, false>(A, 64, B, 0, prm, half * 8, lane, pt0); __syncthreads(); // feat

  { // append de at cols 256..287 of B (view input = [feat(256), de(27), pad])
    const int pt = tid >> 2, q = tid & 3;
    if (q < 2)
#pragma unroll
      for (int ii = 0; ii < 16; ++ii) {
        int f = q * 16 + ii;
        B[pt * LD + 256 + f] = DE[pt * 32 + f];
      }
  }
  __syncthreads();

  mlp_layer<10, 4, true>(B, 0, A, 0, prm, half * 4, lane, pt0); __syncthreads(); // view
  if (half == 0) mlp_head<11, 3, 0>(A, 0, prm, raw, Pbase, lane, pt0);           // rgb
}

// ---------------------------------------------------------------------------
// fp32 params -> fp16 transposed/padded weights [Npad][Kpad] + bias [Npad]
// ---------------------------------------------------------------------------
__global__ void prep_kernel(const float* __restrict__ W, const float* __restrict__ b,
                            _Float16* __restrict__ dst,
                            int Ksrc, int Kpad, int Nsrc, int Npad,
                            int woff, int boff, int skip_remap)
{
  int total = Npad * Kpad;
  for (int i = blockIdx.x * blockDim.x + threadIdx.x; i < total;
       i += gridDim.x * blockDim.x) {
    int n = i / Kpad, k = i % Kpad;
    float v = 0.0f;
    if (n < Nsrc) {
      int ks = k;
      if (skip_remap) { if (k == 63) ks = -1; else if (k > 63) ks = k - 1; }
      if (ks >= 0 && ks < Ksrc) v = W[ks * Nsrc + n];
    }
    dst[woff + n * Kpad + k] = (_Float16)v;
  }
  if (blockIdx.x == 0)
    for (int n = threadIdx.x; n < Npad; n += blockDim.x)
      dst[boff + n] = (_Float16)((n < Nsrc) ? b[n] : 0.0f);
}

// ---------------------------------------------------------------------------
// Coarse weights -> inverse-CDF importance sampling -> merged sorted z_all
// ---------------------------------------------------------------------------
__device__ __forceinline__ float zc_coarse(int i) { return 100.0f * (float)i / 63.0f; }

__global__ void sample_kernel(const float* __restrict__ rays,
                              const float* __restrict__ raw_c,
                              float* __restrict__ z_all)
{
  __shared__ float w[NSC];
  __shared__ float cdf[NSC];   // 63 used
  __shared__ float zm[NSC];    // 63 used
  __shared__ float zs[NSF];
  const int ray = blockIdx.x, tid = threadIdx.x;
  const float* rp = rays + ray * 6;
  float d0 = rp[3], d1 = rp[4], d2 = rp[5];
  float scale = sqrtf(d0*d0 + d1*d1 + d2*d2);

  if (tid == 0) {
    float trans = 1.0f;
    for (int i = 0; i < NSC; ++i) {
      float dist = (i < NSC - 1) ? (zc_coarse(i + 1) - zc_coarse(i)) : 1e10f * scale;
      float sg = raw_c[(ray * NSC + i) * 4 + 3];
      float a = 1.0f - __expf(-fmaxf(sg, 0.0f) * dist);
      w[i] = a * trans;
      trans *= (1.0f - a + 1e-10f);
    }
    float sum = 0.0f;
    for (int i = 1; i < NSC - 1; ++i) sum += w[i] + 1e-5f;
    float c = 0.0f;
    cdf[0] = 0.0f;
    for (int i = 1; i < NSC - 1; ++i) { c += (w[i] + 1e-5f) / sum; cdf[i] = c; }
  }
  if (tid < NSC - 1) zm[tid] = 0.5f * (zc_coarse(tid) + zc_coarse(tid + 1));
  __syncthreads();

  {
    unsigned h = (unsigned)(ray * 9781 + tid * 6271) + 0x9E3779B9u;
    h ^= h >> 16; h *= 0x85EBCA6Bu; h ^= h >> 13; h *= 0xC2B2AE35u; h ^= h >> 16;
    float u = (float)(h >> 8) * (1.0f / 16777216.0f);
    int lo = 0, hi = NSC - 1;                  // searchsorted-right in cdf[0..62]
    while (lo < hi) { int mid = (lo + hi) >> 1; if (cdf[mid] <= u) lo = mid + 1; else hi = mid; }
    int ind = lo;
    int below = (ind - 1 < 0) ? 0 : ind - 1;
    int above = (ind > 62) ? 62 : ind;
    float cb = cdf[below], ca = cdf[above];
    float bb = zm[below],  ba = zm[above];
    float dn = (ca - cb < 1e-5f) ? 1.0f : (ca - cb);
    zs[64 + tid] = bb + (u - cb) / dn * (ba - bb);
    zs[tid] = zc_coarse(tid);
  }
  __syncthreads();

  for (int k = 2; k <= NSF; k <<= 1) {         // bitonic sort of 128 values
    for (int j = k >> 1; j > 0; j >>= 1) {
      for (int i = tid; i < NSF; i += 64) {
        int ixj = i ^ j;
        if (ixj > i) {
          float a = zs[i], b2 = zs[ixj];
          bool up = ((i & k) == 0);
          if ((a > b2) == up) { zs[i] = b2; zs[ixj] = a; }
        }
      }
      __syncthreads();
    }
  }
  for (int i = tid; i < NSF; i += 64) z_all[ray * NSF + i] = zs[i];
}

// ---------------------------------------------------------------------------
// raw2outputs for both fine networks -> concatenated output tuple
// ---------------------------------------------------------------------------
__global__ void out_kernel(const float* __restrict__ rays, const float* __restrict__ z_all,
                           const float* __restrict__ raw_o, const float* __restrict__ raw_s,
                           float* __restrict__ out)
{
  const int ray = blockIdx.x;
  if (threadIdx.x != 0) return;
  const float* rp = rays + ray * 6;
  float d0 = rp[3], d1 = rp[4], d2 = rp[5];
  float scale = sqrtf(d0*d0 + d1*d1 + d2*d2);
  const int SEG = NR * 3 + NR + NR + NR * NSF;   // one net's output block

  for (int net = 0; net < 2; ++net) {
    const float* raw = net ? raw_s : raw_o;
    float* o = out + net * SEG;
    float trans = 1.0f, rm0 = 0, rm1 = 0, rm2 = 0, dep = 0, acc = 0;
    for (int i = 0; i < NSF; ++i) {
      float z = z_all[ray * NSF + i];
      float dist = (i < NSF - 1) ? (z_all[ray * NSF + i + 1] - z) : 1e10f * scale;
      const float* rw = raw + (ray * NSF + i) * 4;
      float s0 = 1.0f / (1.0f + __expf(-rw[0]));
      float s1 = 1.0f / (1.0f + __expf(-rw[1]));
      float s2 = 1.0f / (1.0f + __expf(-rw[2]));
      float alpha = 1.0f - __expf(-fmaxf(rw[3], 0.0f) * dist);
      float wgt = alpha * trans;
      trans *= (1.0f - alpha + 1e-10f);
      rm0 += wgt * s0; rm1 += wgt * s1; rm2 += wgt * s2;
      dep += wgt * (z / scale); acc += wgt;
      o[NR * 3 + NR + NR + ray * NSF + i] = wgt;
    }
    o[ray * 3 + 0] = rm0; o[ray * 3 + 1] = rm1; o[ray * 3 + 2] = rm2;
    o[NR * 3 + ray] = dep;
    o[NR * 3 + NR + ray] = acc;
  }
}

// ---------------------------------------------------------------------------
extern "C" void kernel_launch(void* const* d_in, const int* in_sizes, int n_in,
                              void* d_out, int out_size, void* d_ws, size_t ws_size,
                              hipStream_t stream)
{
  if (n_in < 49) return;
  // Detect flattening: rays (512*6=3072) first (insertion order) or last (sorted).
  int raysIdx = 0, objBase = 1;
  if (in_sizes[0] != NR * 6) { raysIdx = 48; objBase = 0; }
  const int scnBase = objBase + 24;
  // Interleaved (W0,b0,W1,...) vs sorted (W0..W7,Wa,Wf,Wr,Wv,b0..)
  const bool inter = (in_sizes[objBase + 1] == 256);
  static const int WI_int[12] = {0,2,4,6,8,10,12,14,16,18,20,22};
  static const int BI_int[12] = {1,3,5,7,9,11,13,15,17,19,21,23};
  static const int WI_srt[12] = {0,1,2,3,4,5,6,7,8,9,11,10};
  static const int BI_srt[12] = {12,13,14,15,16,17,18,19,20,21,23,22};
  const int* WI = inter ? WI_int : WI_srt;
  const int* BI = inter ? BI_int : BI_srt;

  _Float16* prmO = (_Float16*)d_ws;
  _Float16* prmS = prmO + PARAM_HALVES;
  float* fbase  = (float*)((char*)d_ws + (size_t)2 * PARAM_HALVES * sizeof(_Float16));
  float* raw_c  = fbase;                         // 512*64*4
  float* z_all  = raw_c + NR * NSC * 4;          // 512*128
  float* raw_fo = z_all + NR * NSF;              // 512*128*4
  float* raw_fs = raw_fo + NR * NSF * 4;         // 512*128*4
  const float* rays = (const float*)d_in[raysIdx];

  for (int net = 0; net < 2; ++net) {
    const int base = net ? scnBase : objBase;
    _Float16* dst = net ? prmS : prmO;
    for (int l = 0; l < 12; ++l) {
      const float* W = (const float*)d_in[base + WI[l]];
      const float* b = (const float*)d_in[base + BI[l]];
      prep_kernel<<<64, 256, 0, stream>>>(W, b, dst,
          KSRC_CE[l], KPAD_CE[l], NSRC_CE[l], NPAD_CE[l],
          WOFF_CE[l], BOFF_CE[l], (l == 5) ? 1 : 0);
    }
  }

  nerf_mlp_kernel<<<NR * NSC / 32, 128, 0, stream>>>(rays, prmO, nullptr, NSC, raw_c);
  sample_kernel<<<NR, 64, 0, stream>>>(rays, raw_c, z_all);
  nerf_mlp_kernel<<<NR * NSF / 32, 128, 0, stream>>>(rays, prmO, z_all, NSF, raw_fo);
  nerf_mlp_kernel<<<NR * NSF / 32, 128, 0, stream>>>(rays, prmS, z_all, NSF, raw_fs);
  out_kernel<<<NR, 32, 0, stream>>>(rays, z_all, raw_fo, raw_fs, (float*)d_out);
}
